// TurboGNN_8881992368457
// MI455X (gfx1250) — compile-verified, matrix operand order checked by reading, not run on hardware
//
#include <hip/hip_runtime.h>
#include <hip/hip_bf16.h>
#include <stdint.h>

// MI455X / gfx1250, wave32. WMMA f16->f32 for all dense GEMMs.
// Async global->LDS staging (ASYNCcnt path) when the toolchain exposes it.

typedef __attribute__((ext_vector_type(16))) _Float16 v16h;
typedef __attribute__((ext_vector_type(8)))  float    v8f;

// GLOBAL_LOAD_ASYNC_TO_LDS_B128 builtin takes (v4i addrspace(1)*, v4i addrspace(3)*, Ii, Ii)
typedef int v4i_vs __attribute__((vector_size(16)));
typedef __attribute__((address_space(1))) v4i_vs glb_v4i_t;
typedef __attribute__((address_space(3))) v4i_vs lds_v4i_t;

__device__ __forceinline__ lds_v4i_t* to_lds_v4(void* p) {
    // as3 pointers are 32-bit; low 32 bits of a generic shared ptr = LDS offset
    return (lds_v4i_t*)(uint32_t)(uintptr_t)p;
}
__device__ __forceinline__ glb_v4i_t* to_glb_v4(const void* p) {
    return (glb_v4i_t*)(uintptr_t)p;
}

#if defined(__has_builtin)
#  if __has_builtin(__builtin_amdgcn_global_load_async_to_lds_b128)
#    define HAVE_ASYNC_LDS 1
#  endif
#  if __has_builtin(__builtin_amdgcn_s_wait_asynccnt)
#    define WAIT_ASYNC() __builtin_amdgcn_s_wait_asynccnt(0)
#  endif
#endif
#ifndef HAVE_ASYNC_LDS
#  define HAVE_ASYNC_LDS 0
#endif
#ifndef WAIT_ASYNC
#  define WAIT_ASYNC() asm volatile("s_wait_asynccnt 0" ::: "memory")
#endif

__device__ __forceinline__ float eluf(float v) { return v > 0.f ? v : __expf(v) - 1.f; }

// monotonic float -> uint key (atomicMax-able, order independent => deterministic)
__device__ __forceinline__ uint32_t f2key(float f) {
    uint32_t u = __float_as_uint(f);
    return (u & 0x80000000u) ? ~u : (u | 0x80000000u);
}
__device__ __forceinline__ float key2f(uint32_t k) {
    uint32_t u = (k & 0x80000000u) ? (k ^ 0x80000000u) : ~k;
    return __uint_as_float(u);
}

// ---------------------------------------------------------------------------
// 0) x = elu(LayerNorm(ctrl*w_in + b_in)) * mask  ; also emit f16 copy for WMMA
// ---------------------------------------------------------------------------
__global__ __launch_bounds__(128)
void embed_ln_kernel(const float* __restrict__ ctrl, const float* __restrict__ pmask,
                     const float* __restrict__ w_in, const float* __restrict__ b_in,
                     const float* __restrict__ ln_g, const float* __restrict__ ln_b,
                     float* __restrict__ x, _Float16* __restrict__ xh, int N)
{
    int n = blockIdx.x, d = threadIdx.x;
    float v = ctrl[n] * w_in[d] + b_in[d];
    float s1 = v, s2 = v * v;
    #pragma unroll
    for (int off = 16; off > 0; off >>= 1) {
        s1 += __shfl_xor(s1, off, 32);
        s2 += __shfl_xor(s2, off, 32);
    }
    __shared__ float r1[4], r2[4];
    int wid = d >> 5, lane = d & 31;
    if (lane == 0) { r1[wid] = s1; r2[wid] = s2; }
    __syncthreads();
    float m1 = (r1[0] + r1[1] + r1[2] + r1[3]) * (1.0f / 128.0f);
    float m2 = (r2[0] + r2[1] + r2[2] + r2[3]) * (1.0f / 128.0f);
    float var = m2 - m1 * m1;
    float xn = (v - m1) * rsqrtf(var + 1e-5f) * ln_g[d] + ln_b[d];
    float xe = eluf(xn) * pmask[n];
    x [(size_t)n * 128 + d] = xe;
    xh[(size_t)n * 128 + d] = (_Float16)xe;
}

// ---------------------------------------------------------------------------
// generic f32 -> f16 convert (weights)
// ---------------------------------------------------------------------------
__global__ void f32_to_f16_kernel(const float* __restrict__ in, _Float16* __restrict__ out, int n)
{
    int i = blockIdx.x * blockDim.x + threadIdx.x;
    if (i < n) out[i] = (_Float16)in[i];
}

// ---------------------------------------------------------------------------
// WMMA GEMM: C[M][NC] = A[M][128] * B[128][NC] (+bias). f16 in, f32 accum.
// Block = 128 threads = 4 waves; block tile = 16 rows x 64 cols; each wave
// owns one 16x16 C tile; K=128 = 4 x (16x16x32) v_wmma per wave.
// A tile (contiguous 4KB) staged with GLOBAL_LOAD_ASYNC_TO_LDS_B128 when
// available; B tile staged synchronously (needs the [col][k] transpose so
// fragment reads merge into ds_load_b128).
// ---------------------------------------------------------------------------
template<bool HAS_BIAS, bool OUT_F, bool OUT_H>
__global__ __launch_bounds__(128)
void gemm128_wmma(const _Float16* __restrict__ A,   // [M][128]
                  const _Float16* __restrict__ B,   // [128][NC]
                  const float* __restrict__ bias,   // [NC] (if HAS_BIAS)
                  float* __restrict__ outF,         // [M][NC] (if OUT_F)
                  _Float16* __restrict__ outH,      // [M][NC] (if OUT_H)
                  int M, int NC)
{
    __shared__ _Float16 As[16 * 128];     // [row][k]
    __shared__ _Float16 Bs[64 * 128];     // [col][k]
    const int t = threadIdx.x;
    const int rowBase = blockIdx.x * 16;
    const int colBase = blockIdx.y * 64;
    const bool fullTile = (rowBase + 16) <= M;

    bool asyncA = false;
#if HAVE_ASYNC_LDS
    if (fullTile) {
        // 16 rows x 128 halfs = one contiguous 4KB global region -> LDS
        const _Float16* gsrc = A + (size_t)rowBase * 128;
        __builtin_amdgcn_global_load_async_to_lds_b128(
            to_glb_v4(gsrc + t * 8),        to_lds_v4(As + t * 8),        0, 0);
        __builtin_amdgcn_global_load_async_to_lds_b128(
            to_glb_v4(gsrc + 1024 + t * 8), to_lds_v4(As + 1024 + t * 8), 0, 0);
        asyncA = true;
    }
#endif
    if (!asyncA) {   // synchronous fallback (tail tile or no async support)
        const uint32_t* Ap = (const uint32_t*)A;
        uint32_t* AsW = (uint32_t*)As;
        #pragma unroll
        for (int i = 0; i < 8; ++i) {
            int idx = i * 128 + t;        // dword index in 16x64-dword tile
            int row = idx >> 6, kd = idx & 63;
            int grow = rowBase + row;
            AsW[idx] = (grow < M) ? Ap[(size_t)grow * 64 + kd] : 0u;
        }
    }
    {   // stage B tile: Bs[col][k]; global reads coalesced along columns
        const uint32_t* Bp = (const uint32_t*)B;
        int c2 = t & 31;                  // dword (column pair) within 64 cols
        int k0 = t >> 5;                  // 0..3
        int ncd = NC >> 1;
        #pragma unroll
        for (int kk = 0; kk < 32; ++kk) {
            int k = k0 + kk * 4;
            union { uint32_t u; _Float16 h[2]; } cv;
            cv.u = Bp[(size_t)k * ncd + (colBase >> 1) + c2];
            Bs[(2 * c2)     * 128 + k] = cv.h[0];
            Bs[(2 * c2 + 1) * 128 + k] = cv.h[1];
        }
    }
#if HAVE_ASYNC_LDS
    if (asyncA) WAIT_ASYNC();
#endif
    __syncthreads();

    const int wid  = t >> 5;
    const int lane = t & 31;
    const int g    = lane >> 4;           // lane half-group
    const int nm   = lane & 15;           // m for A rows, n for B/C cols
    const int wc   = wid * 16;            // wave's column slice

    v8f acc = {};
    #pragma unroll
    for (int kc = 0; kc < 4; ++kc) {
        v16h a, b;
        uint32_t* ap = (uint32_t*)&a;
        uint32_t* bp = (uint32_t*)&b;
        #pragma unroll
        for (int v = 0; v < 8; ++v) {
            // ISA 16-bit A 16x32 layout: dword v holds K pair
            int kl = (v < 4) ? (g * 8 + 2 * v) : (16 + g * 8 + 2 * (v - 4));
            int k  = kc * 32 + kl;
            ap[v] = *(const uint32_t*)&As[nm * 128 + k];
            bp[v] = *(const uint32_t*)&Bs[(wc + nm) * 128 + k];
        }
        acc = __builtin_amdgcn_wmma_f32_16x16x32_f16(false, a, false, b,
                                                     (short)0, acc, false, false);
    }

    // C layout: VGPR r -> M = 8*g + r, N = nm
    const int col = colBase + wc + nm;
    const float bcol = HAS_BIAS ? bias[col] : 0.0f;
    size_t o = (size_t)(rowBase + g * 8) * NC + col;
    if (fullTile) {                       // fast path: all 8 stores unconditional
        #pragma unroll
        for (int r = 0; r < 8; ++r) {
            float v = acc[r] + bcol;
            if (OUT_F) outF[o] = v;
            if (OUT_H) outH[o] = (_Float16)v;
            o += NC;
        }
    } else {                              // tail tile: guard each row
        #pragma unroll
        for (int r = 0; r < 8; ++r) {
            int row = rowBase + g * 8 + r;
            if (row < M) {
                float v = acc[r] + bcol;
                if (OUT_F) outF[o] = v;
                if (OUT_H) outH[o] = (_Float16)v;
            }
            o += NC;
        }
    }
}

// ---------------------------------------------------------------------------
// per (node, head): a_s = <h[n,h,:], a_src[h]>, a_d = <h[n,h,:], a_dst[h]>
// ---------------------------------------------------------------------------
__global__ __launch_bounds__(256)
void attn_coeff_kernel(const _Float16* __restrict__ hh,
                       const float* __restrict__ a_src, const float* __restrict__ a_dst,
                       float* __restrict__ asv, float* __restrict__ adv, int N)
{
    int wid = threadIdx.x >> 5, lane = threadIdx.x & 31;
    int n = blockIdx.x, h = wid;
    const _Float16* hp = hh + (size_t)n * 1024 + h * 128;
    float ss = 0.f, sd = 0.f;
    #pragma unroll
    for (int i = 0; i < 4; ++i) {
        int d = lane + 32 * i;
        float hv = (float)hp[d];
        ss += hv * a_src[h * 128 + d];
        sd += hv * a_dst[h * 128 + d];
    }
    #pragma unroll
    for (int off = 16; off > 0; off >>= 1) {
        ss += __shfl_xor(ss, off, 32);
        sd += __shfl_xor(sd, off, 32);
    }
    if (lane == 0) { asv[n * 8 + h] = ss; adv[n * 8 + h] = sd; }
}

__device__ __forceinline__ void edge_ends(const int* ei, int E, int e, int& s, int& d)
{
    if (e < E) { s = ei[e]; d = ei[E + e]; }
    else       { s = d = e - E; }          // appended self loops
}

// pass 1: e = leaky_relu(a_s[src]+a_d[dst]); segment max via atomic key-max
__global__ void edge_logits_kernel(const int* __restrict__ ei, int E, int Etot,
                                   const float* __restrict__ asv, const float* __restrict__ adv,
                                   float* __restrict__ ebuf, uint32_t* __restrict__ mu)
{
    int t = blockIdx.x * blockDim.x + threadIdx.x;
    if (t >= Etot * 8) return;
    int e = t >> 3, h = t & 7, s, d;
    edge_ends(ei, E, e, s, d);
    float ev = asv[s * 8 + h] + adv[d * 8 + h];
    ev = (ev < 0.f) ? 0.2f * ev : ev;
    ebuf[t] = ev;
    atomicMax(&mu[d * 8 + h], f2key(ev));
}

// pass 2a: w = exp(e - max); segment sum via atomicAdd
__global__ void edge_weights_kernel(const int* __restrict__ ei, int E, int Etot,
                                    const float* __restrict__ ebuf, const uint32_t* __restrict__ mu,
                                    float* __restrict__ wbuf, float* __restrict__ ssum)
{
    int t = blockIdx.x * blockDim.x + threadIdx.x;
    if (t >= Etot * 8) return;
    int e = t >> 3, h = t & 7, s, d;
    edge_ends(ei, E, e, s, d);
    float w = __expf(ebuf[t] - key2f(mu[d * 8 + h]));
    wbuf[t] = w;
    atomicAdd(&ssum[d * 8 + h], w);
}

// pass 2b: out[dst] += w * h[src]  (normalization by 1/sum deferred to finalize)
__global__ __launch_bounds__(256)
void edge_aggregate_kernel(const int* __restrict__ ei, int E,
                           const _Float16* __restrict__ hh, const float* __restrict__ wbuf,
                           float* __restrict__ outacc)
{
    int e = blockIdx.x, s, d;
    edge_ends(ei, E, e, s, d);
    const _Float16* hp = hh + (size_t)s * 1024;
    float* op = outacc + (size_t)d * 1024;
    __builtin_prefetch(hp + threadIdx.x * 4, 0, 1);   // global_prefetch_b8 of source row
    #pragma unroll
    for (int i = 0; i < 4; ++i) {
        int idx = i * 256 + threadIdx.x;              // idx = head*128 + dim
        float w = wbuf[e * 8 + (idx >> 7)];
        atomicAdd(op + idx, w * (float)hp[idx]);
    }
}

// per node: x = elu( BN( mean_h(out[h,:]/sum[h]) + bias ) (+res) ); emit f32+f16
__global__ __launch_bounds__(128)
void gat_finalize_kernel(const float* __restrict__ outacc, const float* __restrict__ ssum,
                         const float* __restrict__ bias,
                         const float* __restrict__ bng, const float* __restrict__ bnb,
                         const float* __restrict__ bnrm, const float* __restrict__ bnrv,
                         const float* __restrict__ resid,
                         float* __restrict__ x, _Float16* __restrict__ xh, int N)
{
    int n = blockIdx.x, d = threadIdx.x;
    __shared__ float invs[8];
    if (d < 8) invs[d] = 1.0f / (ssum[n * 8 + d] + 1e-16f);
    __syncthreads();
    const float* op = outacc + (size_t)n * 1024 + d;
    float acc = 0.f;
    #pragma unroll
    for (int h = 0; h < 8; ++h) acc += op[h * 128] * invs[h];
    float v = acc * 0.125f + bias[d];
    v = (v - bnrm[d]) * bng[d] * rsqrtf(bnrv[d] + 1e-5f) + bnb[d];
    if (resid) v += resid[(size_t)n * 128 + d];
    v = eluf(v);
    x [(size_t)n * 128 + d] = v;
    xh[(size_t)n * 128 + d] = (_Float16)v;
}

// head: out[n] = <x[n,:], w_head> + b_head
__global__ __launch_bounds__(256)
void head_kernel(const float* __restrict__ x, const float* __restrict__ w_head,
                 const float* __restrict__ b_head, float* __restrict__ out, int N)
{
    int wid = threadIdx.x >> 5, lane = threadIdx.x & 31;
    int n = blockIdx.x * 8 + wid;
    if (n >= N) return;
    float s = 0.f;
    #pragma unroll
    for (int i = 0; i < 4; ++i) {
        int d = lane + 32 * i;
        s += x[(size_t)n * 128 + d] * w_head[d];
    }
    #pragma unroll
    for (int off = 16; off > 0; off >>= 1) s += __shfl_xor(s, off, 32);
    if (lane == 0) out[n] = s + b_head[0];
}

// ---------------------------------------------------------------------------
// Input layout (setup_inputs dict insertion order, params flattened leaf-wise):
// 0 ctrl_expr[N] 1 perturbation_mask[N] 2 edge_index[2*E]
// 3 w_in 4 b_in 5 ln_g 6 ln_b
// 7+8l: convl.{W,a_src,a_dst,bias}  11+8l: bnl.{g,b,rm,rv}   (l = 0,1,2)
// 31 w_res 32 b_res 33 w_head 34 b_head
// ---------------------------------------------------------------------------
extern "C" void kernel_launch(void* const* d_in, const int* in_sizes, int n_in,
                              void* d_out, int out_size, void* d_ws, size_t ws_size,
                              hipStream_t stream)
{
    const float* ctrl  = (const float*)d_in[0];
    const float* pmask = (const float*)d_in[1];
    const int*   ei    = (const int*)d_in[2];
    const int N = in_sizes[0];
    const int E = in_sizes[2] / 2;
    const int Etot = E + N;

    const float* w_in   = (const float*)d_in[3];
    const float* b_in   = (const float*)d_in[4];
    const float* ln_g   = (const float*)d_in[5];
    const float* ln_b   = (const float*)d_in[6];
    const float* w_res  = (const float*)d_in[31];
    const float* b_res  = (const float*)d_in[32];
    const float* w_head = (const float*)d_in[33];
    const float* b_head = (const float*)d_in[34];

    // workspace carve (~159 MB)
    char* p = (char*)d_ws;
    auto carve = [&](size_t bytes) -> void* {
        void* q = (void*)p; p += (bytes + 255) & ~(size_t)255; return q;
    };
    float*    x      = (float*)   carve((size_t)N * 128 * 4);
    _Float16* xh     = (_Float16*)carve((size_t)N * 128 * 2);
    float*    resb   = (float*)   carve((size_t)N * 128 * 4);
    _Float16* hh     = (_Float16*)carve((size_t)N * 1024 * 2);
    float*    outacc = (float*)   carve((size_t)N * 1024 * 4);
    float*    asv    = (float*)   carve((size_t)N * 8 * 4);
    float*    adv    = (float*)   carve((size_t)N * 8 * 4);
    uint32_t* mu     = (uint32_t*)carve((size_t)N * 8 * 4);
    float*    ssum   = (float*)   carve((size_t)N * 8 * 4);
    float*    ebuf   = (float*)   carve((size_t)Etot * 8 * 4);
    float*    wbuf   = (float*)   carve((size_t)Etot * 8 * 4);
    _Float16* Wh     = (_Float16*)carve((size_t)128 * 1024 * 2);
    _Float16* Wresh  = (_Float16*)carve((size_t)128 * 128 * 2);
    (void)n_in; (void)out_size; (void)ws_size;

    // embed + LN + elu + mask
    embed_ln_kernel<<<N, 128, 0, stream>>>(ctrl, pmask, w_in, b_in, ln_g, ln_b, x, xh, N);

    // residual projection: res = x @ w_res + b_res   (WMMA, f32 out + bias)
    f32_to_f16_kernel<<<(128 * 128 + 255) / 256, 256, 0, stream>>>(w_res, Wresh, 128 * 128);
    gemm128_wmma<true, true, false><<<dim3((N + 15) / 16, 128 / 64), 128, 0, stream>>>(
        xh, Wresh, b_res, resb, nullptr, N, 128);

    for (int l = 0; l < 3; ++l) {
        const float* W    = (const float*)d_in[7  + 8 * l];
        const float* a_s  = (const float*)d_in[8  + 8 * l];
        const float* a_d  = (const float*)d_in[9  + 8 * l];
        const float* bias = (const float*)d_in[10 + 8 * l];
        const float* bng  = (const float*)d_in[11 + 8 * l];
        const float* bnb  = (const float*)d_in[12 + 8 * l];
        const float* bnrm = (const float*)d_in[13 + 8 * l];
        const float* bnrv = (const float*)d_in[14 + 8 * l];

        // h = x @ W  (N x 1024), f16 output for cheap gathers + WMMA reuse
        f32_to_f16_kernel<<<(128 * 1024 + 255) / 256, 256, 0, stream>>>(W, Wh, 128 * 1024);
        gemm128_wmma<false, false, true><<<dim3((N + 15) / 16, 1024 / 64), 128, 0, stream>>>(
            xh, Wh, nullptr, nullptr, hh, N, 1024);

        attn_coeff_kernel<<<N, 256, 0, stream>>>(hh, a_s, a_d, asv, adv, N);

        (void)hipMemsetAsync(mu,     0, (size_t)N * 8 * 4,    stream);
        (void)hipMemsetAsync(ssum,   0, (size_t)N * 8 * 4,    stream);
        (void)hipMemsetAsync(outacc, 0, (size_t)N * 1024 * 4, stream);

        int nthr = Etot * 8;
        edge_logits_kernel <<<(nthr + 255) / 256, 256, 0, stream>>>(ei, E, Etot, asv, adv, ebuf, mu);
        edge_weights_kernel<<<(nthr + 255) / 256, 256, 0, stream>>>(ei, E, Etot, ebuf, mu, wbuf, ssum);
        edge_aggregate_kernel<<<Etot, 256, 0, stream>>>(ei, E, hh, wbuf, outacc);

        gat_finalize_kernel<<<N, 128, 0, stream>>>(outacc, ssum, bias, bng, bnb, bnrm, bnrv,
                                                   (l == 2) ? resb : nullptr, x, xh, N);
    }

    head_kernel<<<(N + 7) / 8, 256, 0, stream>>>(x, w_head, b_head, (float*)d_out, N);
}
